// WReN_27006754357644
// MI455X (gfx1250) — matrix operand order repaired and checked by reading, not compiled, not run on hardware
//
#include <hip/hip_runtime.h>
#include <hip/hip_bf16.h>

// ---------------- WMMA plumbing (gfx1250, wave32) ----------------
typedef __attribute__((ext_vector_type(16))) __bf16 v16bf;
typedef __attribute__((ext_vector_type(8)))  float  v8f;

union ABf16 { uint4 u[2]; v16bf v; };

__device__ __forceinline__ unsigned short f2bf(float f) {
    unsigned int u = __float_as_uint(f);
    u += 0x7FFFu + ((u >> 16) & 1u);        // round-to-nearest-even
    return (unsigned short)(u >> 16);
}

__device__ __forceinline__ v8f wmma_bf16(uint4 a0, uint4 a1, uint4 b0, uint4 b1, v8f c) {
    ABf16 a, b;
    a.u[0] = a0; a.u[1] = a1;
    b.u[0] = b0; b.u[1] = b1;
    return __builtin_amdgcn_wmma_f32_16x16x32_bf16(false, a.v, false, b.v,
                                                   (short)0, c, false, false);
}

// ---------------- Kernel 0: transpose + fp32->bf16 weights ----------------
// z=0: Wg1 top (rows 0..511)  z=1: Wg1 bottom  z=2: Wg2  z=3: Wf1
// output WT[c*512 + k] = bf16(W[k*512 + c])   (N x K, row-major)
__global__ __launch_bounds__(256)
void wt_transpose_kernel(const float* __restrict__ Wg1,
                         const float* __restrict__ Wg2,
                         const float* __restrict__ Wf1,
                         unsigned short* o0, unsigned short* o1,
                         unsigned short* o2, unsigned short* o3) {
    int z = blockIdx.z;
    int o = blockIdx.x * 256 + threadIdx.x;     // 0 .. 512*512-1
    int c = o >> 9;
    int k = o & 511;
    const float* src;
    unsigned short* dst;
    switch (z) {
        case 0:  src = Wg1;               dst = o0; break;
        case 1:  src = Wg1 + 512 * 512;   dst = o1; break;
        case 2:  src = Wg2;               dst = o2; break;
        default: src = Wf1;               dst = o3; break;
    }
    dst[o] = f2bf(src[(size_t)k * 512 + c]);
}

// ---------------- Generic WMMA GEMM: out = act(A @ B^T_bf16 + bias) --------
// A rows: gm < Msplit -> A0[gm], else A1[gm - Msplit].  K = N = 512.
// grid: (M/16, 4, nHalves). block 256 = 8 waves, wave owns one 16-col tile.
__global__ __launch_bounds__(256)
void gemm512_bf16_kernel(const float* __restrict__ A0, const float* __restrict__ A1,
                         int Msplit, int M,
                         const unsigned short* __restrict__ BT0,
                         const unsigned short* __restrict__ BT1,
                         float* __restrict__ out0, float* __restrict__ out1,
                         const float* __restrict__ bias, int doRelu) {
    __shared__ unsigned short aT[16 * 512];     // 16 KB bf16 A tile

    const int tid   = threadIdx.x;
    const int lane  = tid & 31;
    const int wave  = tid >> 5;
    const int mtile = blockIdx.x;
    const int half  = blockIdx.z;
    const unsigned short* BT = half ? BT1 : BT0;
    float* out = half ? out1 : out0;

    // ---- stage A tile (fp32 -> bf16 in LDS) ----
    {
        int r  = tid >> 4;              // row 0..15
        int c0 = (tid & 15) * 32;       // 32 cols per thread
        int gm = mtile * 16 + r;
        const float* src = (gm < Msplit) ? (A0 + (size_t)gm * 512 + c0)
                                         : (A1 + (size_t)(gm - Msplit) * 512 + c0);
        #pragma unroll
        for (int i = 0; i < 8; ++i) {
            float4 f = ((const float4*)src)[i];
            ushort4 h;
            h.x = f2bf(f.x); h.y = f2bf(f.y); h.z = f2bf(f.z); h.w = f2bf(f.w);
            ((ushort4*)&aT[r * 512 + c0])[i] = h;
        }
    }
    __syncthreads();

    const int ntile = blockIdx.y * 8 + wave;    // 0..31
    const int n     = lane & 15;
    const int colg  = ntile * 16 + n;
    const int mloc  = lane & 15;
    const int kbA   = (lane < 16) ? 0 : 16;     // A: K 0-15 vs 16-31
    const int kbB   = (lane < 16) ? 0 : 16;     // B: K 0-15 vs 16-31

    v8f c = {};
    #pragma unroll
    for (int kq = 0; kq < 16; ++kq) {
        const unsigned short* ap = &aT[mloc * 512 + kq * 32 + kbA];
        uint4 a0 = *(const uint4*)ap;
        uint4 a1 = *(const uint4*)(ap + 8);
        const unsigned short* bp = BT + (size_t)colg * 512 + kq * 32 + kbB;
        __builtin_prefetch((const void*)(bp + 32), 0, 3);
        uint4 b0 = *(const uint4*)bp;
        uint4 b1 = *(const uint4*)(bp + 8);
        c = wmma_bf16(a0, a1, b0, b1, c);
    }

    // ---- epilogue: bias, relu, store fp32 ----
    float bv = bias ? bias[colg] : 0.0f;
    int mbase = (lane < 16) ? 0 : 8;
    #pragma unroll
    for (int v = 0; v < 8; ++v) {
        int gm = mtile * 16 + mbase + v;
        float val = c[v] + bv;
        if (doRelu) val = fmaxf(val, 0.0f);
        if (gm < M) out[(size_t)gm * 512 + colg] = val;
    }
}

// ---------------- Kernel 2: fused layer1-combine + layer2 GEMM + pair sum --
// ctx = b*2 + way (128 ctx).  Per ctx: 18 M-tiles of u rows:
//   t=0,1 : 30 shot-shot pairs (shared across queries)
//   t=2+q : 12 query pairs for query q
// s[(b*16+q)*2+way, col] = ss_col + qsum_col
__global__ __launch_bounds__(256)
void relnet_mid_kernel(const float* __restrict__ XJ, const float* __restrict__ XI,
                       const unsigned short* __restrict__ Wg2T,
                       const float* __restrict__ bg1, const float* __restrict__ bg2,
                       float* __restrict__ S) {
    __shared__ unsigned short uT[16 * 512];

    const int tid  = threadIdx.x;
    const int lane = tid & 31;
    const int wave = tid >> 5;
    const int ctx  = blockIdx.x;            // 0..127
    const int b    = ctx >> 1;
    const int way  = ctx & 1;
    const int sb    = b * 12 + way * 6;     // shot row base in X_all
    const int qbase = 768 + b * 16;         // query row base in X_all

    const int ntile = blockIdx.y * 8 + wave;
    const int n     = lane & 15;
    const int colg  = ntile * 16 + n;
    const int mloc  = lane & 15;
    const int kbA   = (lane < 16) ? 0 : 16;
    const int kbB   = (lane < 16) ? 0 : 16;
    const int mbase = (lane < 16) ? 0 : 8;

    // ---- cache this wave's B k-panel in VGPRs (16 k-steps x 32B) ----
    uint4 bR[32];
    #pragma unroll
    for (int kq = 0; kq < 16; ++kq) {
        const unsigned short* bp = Wg2T + (size_t)colg * 512 + kq * 32 + kbB;
        bR[2 * kq]     = *(const uint4*)bp;
        bR[2 * kq + 1] = *(const uint4*)(bp + 8);
    }
    const float b2 = bg2[colg];

    float ss = 0.0f;                         // shot-shot column sum (per lane)

    for (int t = 0; t < 18; ++t) {
        __syncthreads();
        // ---- stage u tile: u = relu(XJ[jrow] + XI[irow] + bg1) as bf16 ----
        {
            int r  = tid >> 4;
            int c0 = (tid & 15) * 32;
            int jrow, irow;
            bool valid;
            if (t < 2) {
                int p = t * 16 + r;                     // pair index 0..29
                valid = (p < 30);
                int i  = p / 5;
                int jj = p % 5;
                int j  = jj + (jj >= i ? 1 : 0);
                if (!valid) { i = 0; j = 0; }
                jrow = sb + j; irow = sb + i;
            } else {
                int q = t - 2;
                int qrow = qbase + q;
                valid = (r < 12);
                if (r < 6)        { jrow = qrow;         irow = sb + r; }
                else if (r < 12)  { jrow = sb + (r - 6); irow = qrow;   }
                else              { jrow = qrow;         irow = qrow;   }
            }
            const float* pj = XJ + (size_t)jrow * 512 + c0;
            const float* pi = XI + (size_t)irow * 512 + c0;
            const float* pb = bg1 + c0;
            #pragma unroll
            for (int i2 = 0; i2 < 8; ++i2) {
                float4 fj = ((const float4*)pj)[i2];
                float4 fi = ((const float4*)pi)[i2];
                float4 fb = ((const float4*)pb)[i2];
                float ux = fmaxf(fj.x + fi.x + fb.x, 0.0f);
                float uy = fmaxf(fj.y + fi.y + fb.y, 0.0f);
                float uz = fmaxf(fj.z + fi.z + fb.z, 0.0f);
                float uw = fmaxf(fj.w + fi.w + fb.w, 0.0f);
                if (!valid) { ux = uy = uz = uw = 0.0f; }
                ushort4 h;
                h.x = f2bf(ux); h.y = f2bf(uy); h.z = f2bf(uz); h.w = f2bf(uw);
                ((ushort4*)&uT[r * 512 + c0])[i2] = h;
            }
        }
        __syncthreads();

        // ---- 16x16x512 via 16 WMMAs, B from registers ----
        v8f c = {};
        #pragma unroll
        for (int kq = 0; kq < 16; ++kq) {
            const unsigned short* ap = &uT[mloc * 512 + kq * 32 + kbA];
            uint4 a0 = *(const uint4*)ap;
            uint4 a1 = *(const uint4*)(ap + 8);
            c = wmma_bf16(a0, a1, bR[2 * kq], bR[2 * kq + 1], c);
        }

        // ---- epilogue: relu(+bg2), masked row-sum within tile ----
        int vr = (t == 0) ? 16 : (t == 1) ? 14 : 12;   // valid rows this tile
        float part = 0.0f;
        #pragma unroll
        for (int v = 0; v < 8; ++v) {
            int m = mbase + v;
            float val = fmaxf(c[v] + b2, 0.0f);
            part += (m < vr) ? val : 0.0f;
        }
        part += __shfl_xor(part, 16, 32);   // combine rows 0-7 with 8-15 halves

        if (t < 2) {
            ss += part;                      // shot-shot, finished after t=1
        } else {
            int q = t - 2;
            if (lane < 16) {
                size_t row = (size_t)(b * 16 + q) * 2 + way;
                S[row * 512 + colg] = ss + part;
            }
        }
    }
}

// ---------------- Kernel 4: logits = T @ Wf2 + bf2 (wave-per-row dot) ------
__global__ __launch_bounds__(256)
void final_logit_kernel(const float* __restrict__ T, const float* __restrict__ Wf2,
                        const float* __restrict__ bf2, float* __restrict__ out) {
    int wave = threadIdx.x >> 5;
    int lane = threadIdx.x & 31;
    int row  = blockIdx.x * 8 + wave;       // 0..2047
    const float* t = T + (size_t)row * 512;
    float acc = 0.0f;
    #pragma unroll
    for (int i = 0; i < 16; ++i) {
        int c = i * 32 + lane;
        acc += t[c] * Wf2[c];
    }
    #pragma unroll
    for (int d = 16; d >= 1; d >>= 1) acc += __shfl_xor(acc, d, 32);
    if (lane == 0) out[row] = acc + bf2[0];
}

// ---------------- Host launcher --------------------------------------------
extern "C" void kernel_launch(void* const* d_in, const int* in_sizes, int n_in,
                              void* d_out, int out_size, void* d_ws, size_t ws_size,
                              hipStream_t stream) {
    const float* x_shot  = (const float*)d_in[0];   // 768 x 512
    const float* x_query = (const float*)d_in[1];   // 1024 x 512
    const float* Wg1 = (const float*)d_in[2];       // 1024 x 512
    const float* bg1 = (const float*)d_in[3];
    const float* Wg2 = (const float*)d_in[4];       // 512 x 512
    const float* bg2 = (const float*)d_in[5];
    const float* Wf1 = (const float*)d_in[6];       // 512 x 512
    const float* bf1 = (const float*)d_in[7];
    const float* Wf2 = (const float*)d_in[8];       // 512 x 1
    const float* bf2 = (const float*)d_in[9];
    float* out = (float*)d_out;                     // 2048 fp32 (b, q, way)

    char* ws = (char*)d_ws;
    size_t off = 0;
    auto carve = [&](size_t bytes) -> char* {
        char* p = ws + off;
        off += (bytes + 255) & ~(size_t)255;
        return p;
    };
    unsigned short* wg1tT = (unsigned short*)carve(512 * 512 * 2);
    unsigned short* wg1tB = (unsigned short*)carve(512 * 512 * 2);
    unsigned short* wg2T  = (unsigned short*)carve(512 * 512 * 2);
    unsigned short* wf1T  = (unsigned short*)carve(512 * 512 * 2);
    float* XJ = (float*)carve((size_t)1792 * 512 * 4);
    float* XI = (float*)carve((size_t)1792 * 512 * 4);
    float* S  = (float*)carve((size_t)2048 * 512 * 4);
    float* T  = (float*)carve((size_t)2048 * 512 * 4);

    // 0) weight transpose + bf16 convert
    wt_transpose_kernel<<<dim3(1024, 1, 4), 256, 0, stream>>>(
        Wg1, Wg2, Wf1, wg1tT, wg1tB, wg2T, wf1T);

    // 1) XJ = X_all @ Wg1_top ; XI = X_all @ Wg1_bot   (M = 1792)
    gemm512_bf16_kernel<<<dim3(112, 4, 2), 256, 0, stream>>>(
        x_shot, x_query, 768, 1792, wg1tT, wg1tB, XJ, XI, nullptr, 0);

    // 2) fused relu(XJ+XI+bg1) @ Wg2 -> relu(+bg2) -> pair sums -> S (2048x512)
    relnet_mid_kernel<<<dim3(128, 4, 1), 256, 0, stream>>>(
        XJ, XI, wg2T, bg1, bg2, S);

    // 3) T = relu(S @ Wf1 + bf1)   (M = 2048)
    gemm512_bf16_kernel<<<dim3(128, 4, 1), 256, 0, stream>>>(
        S, S, 4096, 2048, wf1T, wf1T, T, T, bf1, 1);

    // 4) out = T @ Wf2 + bf2
    final_logit_kernel<<<dim3(256, 1, 1), 256, 0, stream>>>(T, Wf2, bf2, out);
}
